// MyViT_25984552141005
// MI455X (gfx1250) — compile-verified, matrix operand order checked by reading, not run on hardware
//
#include <hip/hip_runtime.h>
#include <hip/hip_bf16.h>
#include <math.h>

typedef __bf16 bf16;
typedef __attribute__((ext_vector_type(16))) __bf16 v16bf;
typedef __attribute__((ext_vector_type(8)))  float  v8f;
typedef __attribute__((ext_vector_type(4)))  int    v4i;

#define N_IMG   256
#define S_LEN   65
#define D_MODEL 768
#define NHEADS  12
#define HD      64
#define L_LAYERS 12
#define MLP_D   3072
#define M_ROWS  (N_IMG * S_LEN)   // 16640, multiple of 128
#define OUT_N   1000
#define OUT_PAD 1024

// ---------------- fragment helpers (CDNA5 WMMA bf16 layouts) ----------------
union FragB { v16bf v; uint4 q[2]; bf16 e[16]; };
union FAcc  { v8f v; float f[8]; };

// A (16x32, 16-bit): lane<16 holds row M=lane, K = {0..7, 16..23}; lane>=16 row M=lane-16,
// K = {8..15, 24..31}. Per-lane data = two contiguous 16B chunks at kb and kb+16.
// B (32x16) mirrors A with N as the lane dimension; with a row-major weight W[nout][k],
// lane's column nout reads contiguously along k -> same loader.
__device__ inline v16bf frag_load(const bf16* p) {
    FragB f;
    f.q[0] = *(const uint4*)(p);
    f.q[1] = *(const uint4*)(p + 16);
    return f.v;
}
__device__ inline v16bf frag_zero() {
    FragB f;
    f.q[0] = make_uint4(0u, 0u, 0u, 0u);
    f.q[1] = make_uint4(0u, 0u, 0u, 0u);
    return f.v;
}
__device__ inline void facc_zero(FAcc& a) {
#pragma unroll
    for (int i = 0; i < 8; ++i) a.f[i] = 0.0f;
}
__device__ inline v8f wmma_bf16(v16bf a, v16bf b, v8f c) {
    return __builtin_amdgcn_wmma_f32_16x16x32_bf16(false, a, false, b, (short)0, c, false, false);
}

// ---------------- async global->LDS copy (CDNA5), with safe fallback ----------------
#if __has_builtin(__builtin_amdgcn_global_load_async_to_lds_b128)
#define HAVE_ASYNC_LDS 1
__device__ inline void async_copy16(const bf16* g, bf16* l) {
    __builtin_amdgcn_global_load_async_to_lds_b128(
        (__attribute__((address_space(1))) v4i*)(g),
        (__attribute__((address_space(3))) v4i*)(l), 0, 0);
}
#else
#define HAVE_ASYNC_LDS 0
__device__ inline void async_copy16(const bf16* g, bf16* l) {
    *(uint4*)l = *(const uint4*)g;
}
#endif
__device__ inline void async_wait() {
#if __has_builtin(__builtin_amdgcn_s_wait_asynccnt)
    __builtin_amdgcn_s_wait_asynccnt(0);
#endif
}

// ---------------- elementwise / conversion kernels ----------------
__global__ void cvt_bf16_kernel(const float* __restrict__ s, bf16* __restrict__ d, long n) {
    long i = (long)blockIdx.x * blockDim.x + threadIdx.x;
    if (i < n) d[i] = (bf16)s[i];
}

__global__ void cvt_pad_rows_kernel(const float* __restrict__ s, bf16* __restrict__ d,
                                    int rows_valid, int rows_pad, int cols) {
    long i = (long)blockIdx.x * blockDim.x + threadIdx.x;
    if (i >= (long)rows_pad * cols) return;
    int r = (int)(i / cols), c = (int)(i % cols);
    d[i] = (r < rows_valid) ? (bf16)s[(long)r * cols + c] : (bf16)0.0f;
}

// patch embed + cls token + sinusoidal positions (faithful: uses raw index i)
__global__ void patch_embed_kernel(const float* __restrict__ img, const float* __restrict__ Wp,
                                   const float* __restrict__ bp, const float* __restrict__ cls,
                                   float* __restrict__ x) {
    long i = (long)blockIdx.x * blockDim.x + threadIdx.x;
    if (i >= (long)M_ROWS * D_MODEL) return;
    int d = (int)(i % D_MODEL);
    int m = (int)(i / D_MODEL);
    int n = m / S_LEN, s = m % S_LEN;
    float val;
    if (s == 0) {
        val = cls[d];
    } else {
        int p = s - 1;
        val = bp[d];
        const float* ib = img + (long)n * 3 * 1024 + p * 16;
        const float* wb = Wp + (long)d * 48;
#pragma unroll
        for (int c = 0; c < 3; ++c)
#pragma unroll
            for (int j = 0; j < 16; ++j)
                val += ib[c * 1024 + j] * wb[c * 16 + j];
    }
    double ang = (double)s / pow(10000.0, 2.0 * (double)d / 768.0);
    float pe = (d & 1) ? (float)cos(ang) : (float)sin(ang);
    x[i] = val + pe;
}

// row LayerNorm (fp32 in, bf16 out), one 256-thread block per row of 768
__global__ void layernorm_kernel(const float* __restrict__ x, const float* __restrict__ g,
                                 const float* __restrict__ b, bf16* __restrict__ out) {
    __shared__ float red[256];
    int row = blockIdx.x, tid = threadIdx.x;
    const float* xr = x + (long)row * D_MODEL;
    float s = 0.0f;
    for (int c = tid; c < D_MODEL; c += 256) s += xr[c];
    red[tid] = s; __syncthreads();
    for (int off = 128; off; off >>= 1) { if (tid < off) red[tid] += red[tid + off]; __syncthreads(); }
    float mean = red[0] * (1.0f / D_MODEL);
    __syncthreads();
    float s2 = 0.0f;
    for (int c = tid; c < D_MODEL; c += 256) { float dl = xr[c] - mean; s2 += dl * dl; }
    red[tid] = s2; __syncthreads();
    for (int off = 128; off; off >>= 1) { if (tid < off) red[tid] += red[tid + off]; __syncthreads(); }
    float inv = rsqrtf(red[0] * (1.0f / D_MODEL) + 1e-5f);
    for (int c = tid; c < D_MODEL; c += 256)
        out[(long)row * D_MODEL + c] = (bf16)((xr[c] - mean) * inv * g[c] + b[c]);
}

// ---------------- per-head QKV projection via WMMA ----------------
// grid (M_ROWS/16, NHEADS), block 128 (4 waves, one 16-wide out-col tile each)
__global__ void qkv_wmma_kernel(const bf16* __restrict__ h,
                                const bf16* __restrict__ wq, const bf16* __restrict__ wk,
                                const bf16* __restrict__ wv,
                                const float* __restrict__ bq, const float* __restrict__ bk,
                                const float* __restrict__ bv,
                                bf16* __restrict__ qo, bf16* __restrict__ ko, bf16* __restrict__ vo) {
    int mt = blockIdx.x, head = blockIdx.y;
    int wave = threadIdx.x >> 5, lane = threadIdx.x & 31;
    int r = lane & 15, kb = (lane >> 4) * 8;
    int m0 = mt * 16, e0 = wave * 16;

    const bf16* W[3]; W[0] = wq + head * HD * HD; W[1] = wk + head * HD * HD; W[2] = wv + head * HD * HD;
    const float* Bv[3]; Bv[0] = bq + head * HD; Bv[1] = bk + head * HD; Bv[2] = bv + head * HD;
    bf16* O[3]; O[0] = qo; O[1] = ko; O[2] = vo;

    const bf16* ap = h + (long)(m0 + r) * D_MODEL + head * HD + kb;
#pragma unroll
    for (int j = 0; j < 3; ++j) {
        FAcc acc; facc_zero(acc);
#pragma unroll
        for (int kt = 0; kt < HD; kt += 32) {
            v16bf af = frag_load(ap + kt);
            v16bf bfr = frag_load(W[j] + (long)(e0 + r) * HD + kt + kb);
            acc.v = wmma_bf16(af, bfr, acc.v);
        }
        int e = e0 + r;
        float bias = Bv[j][e];
#pragma unroll
        for (int t = 0; t < 8; ++t) {
            int m = m0 + t + ((lane >> 4) << 3);
            int n = m / S_LEN, s = m % S_LEN;
            O[j][(((long)n * NHEADS + head) * S_LEN + s) * HD + e] = (bf16)(acc.f[t] + bias);
        }
    }
}

// ---------------- attention: one block per (image, head) ----------------
__global__ void attn_wmma_kernel(const bf16* __restrict__ qb, const bf16* __restrict__ kbuf,
                                 const bf16* __restrict__ vbuf, float* __restrict__ x) {
    __shared__ bf16  ks[80][64];   // K, rows padded to 80
    __shared__ bf16  vT[64][96];   // V transposed, K padded to 96
    __shared__ float sc[80][80];   // scores fp32
    __shared__ bf16  pr[80][96];   // probs bf16, K padded
    int n = blockIdx.x / NHEADS, head = blockIdx.x % NHEADS;
    int tid = threadIdx.x, wave = tid >> 5, lane = tid & 31;
    int r = lane & 15, kb = (lane >> 4) * 8;

    const bf16* qg = qb   + ((long)(n * NHEADS + head)) * S_LEN * HD;
    const bf16* kg = kbuf + ((long)(n * NHEADS + head)) * S_LEN * HD;
    const bf16* vg = vbuf + ((long)(n * NHEADS + head)) * S_LEN * HD;

    // stage K rows 0..64 via async 16B chunks (8 chunks per 64-element row)
    for (int i = tid; i < S_LEN * 8; i += 256) {
        int t = i >> 3, c = (i & 7) * 8;
        async_copy16(kg + t * HD + c, &ks[t][c]);
    }
    // zero pad rows 65..79
    for (int i = tid; i < (80 - S_LEN) * 64; i += 256) {
        int t = S_LEN + i / 64, e = i % 64;
        ks[t][e] = (bf16)0.0f;
    }
    // V transposed (manual: needs element transpose), K padded to 96
    for (int i = tid; i < 64 * 96; i += 256) {
        int e = i / 96, t = i % 96;
        vT[e][t] = (t < S_LEN) ? vg[t * HD + e] : (bf16)0.0f;
    }
    async_wait();
    __syncthreads();

    // scores = q @ k^T * scale : 5x5 tiles of 16x16, K=64
    for (int ti = wave; ti < 25; ti += 8) {
        int mt = ti / 5, nt = ti % 5;
        FAcc acc; facc_zero(acc);
#pragma unroll
        for (int kt = 0; kt < HD; kt += 32) {
            int row = mt * 16 + r;
            v16bf af = (row < S_LEN) ? frag_load(qg + (long)row * HD + kt + kb) : frag_zero();
            v16bf bfr = frag_load(&ks[nt * 16 + r][kt + kb]);
            acc.v = wmma_bf16(af, bfr, acc.v);
        }
#pragma unroll
        for (int t = 0; t < 8; ++t) {
            int m = mt * 16 + t + ((lane >> 4) << 3);
            sc[m][nt * 16 + r] = acc.f[t] * 0.125f;   // 1/sqrt(64)
        }
    }
    __syncthreads();

    // softmax rows (fp32), write bf16 probs with zero padding
    if (tid < 80) {
        int row = tid;
        if (row < S_LEN) {
            float mx = -1e30f;
            for (int t = 0; t < S_LEN; ++t) mx = fmaxf(mx, sc[row][t]);
            float sum = 0.0f;
            for (int t = 0; t < S_LEN; ++t) { float e = __expf(sc[row][t] - mx); sc[row][t] = e; sum += e; }
            float inv = 1.0f / sum;
            for (int t = 0; t < S_LEN; ++t) pr[row][t] = (bf16)(sc[row][t] * inv);
            for (int t = S_LEN; t < 96; ++t) pr[row][t] = (bf16)0.0f;
        } else {
            for (int t = 0; t < 96; ++t) pr[row][t] = (bf16)0.0f;
        }
    }
    __syncthreads();

    // o = probs @ v : 5x4 tiles, K=96, residual-add into x
    for (int ti = wave; ti < 20; ti += 8) {
        int mt = ti / 4, et = ti % 4;
        FAcc acc; facc_zero(acc);
#pragma unroll
        for (int kt = 0; kt < 96; kt += 32) {
            v16bf af  = frag_load(&pr[mt * 16 + r][kt + kb]);
            v16bf bfr = frag_load(&vT[et * 16 + r][kt + kb]);
            acc.v = wmma_bf16(af, bfr, acc.v);
        }
#pragma unroll
        for (int t = 0; t < 8; ++t) {
            int s = mt * 16 + t + ((lane >> 4) << 3);
            if (s < S_LEN) {
                long o = ((long)(n * S_LEN + s)) * D_MODEL + head * HD + et * 16 + r;
                x[o] += acc.f[t];
            }
        }
    }
}

// ---------------- WMMA GEMM with async-LDS-staged B, double buffered ----------------
// out = A[M,K] @ B[N,K]^T + bias. Block 256 = 8 waves, tile 128x64.
// B k-tile (64x32) staged to LDS once per block via GLOBAL_LOAD_ASYNC_TO_LDS_B128,
// prefetched one k-step ahead while WMMAs consume the current buffer.
__global__ void gemm_wmma_kernel(const bf16* __restrict__ A, const bf16* __restrict__ Bw,
                                 const float* __restrict__ bias,
                                 float* resid, float* outf, bf16* outb,
                                 int Nvalid, int K, int ldo, int dogelu) {
    __shared__ bf16 Bs[2][64][32];   // 2 x 4KB
    int tid = threadIdx.x;
    int wave = tid >> 5, lane = tid & 31;
    int r = lane & 15, kb = (lane >> 4) * 8;
    int m0 = blockIdx.x * 128 + wave * 16;
    int n0 = blockIdx.y * 64;

    // staging assignment: one 16B chunk per thread (64 rows x 4 chunks)
    int srow = tid >> 2;
    int skc  = (tid & 3) * 8;
    const bf16* sg = Bw + (long)(n0 + srow) * K + skc;

    FAcc acc[4];
#pragma unroll
    for (int c = 0; c < 4; ++c) facc_zero(acc[c]);

    const bf16* ap = A + (long)(m0 + r) * K + kb;

    // prologue: stage k-tile 0
    async_copy16(sg, &Bs[0][srow][skc]);
    async_wait();
    __syncthreads();

    int buf = 0;
    for (int kt = 0; kt < K; kt += 32) {
        if (kt + 32 < K)
            async_copy16(sg + kt + 32, &Bs[buf ^ 1][srow][skc]);   // prefetch next tile
        v16bf af = frag_load(ap + kt);
        __builtin_prefetch(ap + kt + 256, 0, 0);                   // global_prefetch_b8
#pragma unroll
        for (int c = 0; c < 4; ++c) {
            v16bf bfr = frag_load(&Bs[buf][c * 16 + r][kb]);
            acc[c].v = wmma_bf16(af, bfr, acc[c].v);
        }
        async_wait();
        __syncthreads();
        buf ^= 1;
    }

#pragma unroll
    for (int c = 0; c < 4; ++c) {
        int nn = n0 + c * 16 + r;
        if (nn >= Nvalid) continue;
        float bv = bias ? bias[nn] : 0.0f;
#pragma unroll
        for (int t = 0; t < 8; ++t) {
            int m = m0 + t + ((lane >> 4) << 3);
            float v = acc[c].f[t] + bv;
            if (dogelu) v = 0.5f * v * (1.0f + erff(v * 0.70710678118654752f));
            long o = (long)m * ldo + nn;
            if (resid) { v += resid[o]; resid[o] = v; }
            if (outf) outf[o] = v;
            if (outb) outb[o] = (bf16)v;
        }
    }
}

__global__ void extract_cls_kernel(const float* __restrict__ x, bf16* __restrict__ o) {
    int i = blockIdx.x * blockDim.x + threadIdx.x;
    if (i < N_IMG * D_MODEL) {
        int n = i / D_MODEL, d = i % D_MODEL;
        o[i] = (bf16)x[((long)n * S_LEN) * D_MODEL + d];
    }
}

__global__ void softmax_out_kernel(const float* __restrict__ logits, float* __restrict__ out) {
    __shared__ float red[256];
    int row = blockIdx.x, tid = threadIdx.x;
    const float* lr = logits + (long)row * OUT_PAD;
    float mx = -1e30f;
    for (int c = tid; c < OUT_N; c += 256) mx = fmaxf(mx, lr[c]);
    red[tid] = mx; __syncthreads();
    for (int off = 128; off; off >>= 1) { if (tid < off) red[tid] = fmaxf(red[tid], red[tid + off]); __syncthreads(); }
    mx = red[0]; __syncthreads();
    float s = 0.0f;
    for (int c = tid; c < OUT_N; c += 256) s += expf(lr[c] - mx);
    red[tid] = s; __syncthreads();
    for (int off = 128; off; off >>= 1) { if (tid < off) red[tid] += red[tid + off]; __syncthreads(); }
    float inv = 1.0f / red[0];
    for (int c = tid; c < OUT_N; c += 256)
        out[(long)row * OUT_N + c] = expf(lr[c] - mx) * inv;
}

// ---------------- host orchestration ----------------
extern "C" void kernel_launch(void* const* d_in, const int* in_sizes, int n_in,
                              void* d_out, int out_size, void* d_ws, size_t ws_size,
                              hipStream_t stream) {
    (void)in_sizes; (void)n_in; (void)out_size; (void)ws_size;
    const float* images = (const float*)d_in[0];
    const float* Wp     = (const float*)d_in[1];
    const float* bp     = (const float*)d_in[2];
    const float* cls    = (const float*)d_in[3];
    const float* ln1_g  = (const float*)d_in[4];
    const float* ln1_b  = (const float*)d_in[5];
    const float* Wq     = (const float*)d_in[6];
    const float* bq     = (const float*)d_in[7];
    const float* Wk     = (const float*)d_in[8];
    const float* bk     = (const float*)d_in[9];
    const float* Wv     = (const float*)d_in[10];
    const float* bv     = (const float*)d_in[11];
    const float* ln2_g  = (const float*)d_in[12];
    const float* ln2_b  = (const float*)d_in[13];
    const float* W1     = (const float*)d_in[14];
    const float* b1     = (const float*)d_in[15];
    const float* W2     = (const float*)d_in[16];
    const float* b2     = (const float*)d_in[17];
    const float* Wo     = (const float*)d_in[18];
    const float* bo     = (const float*)d_in[19];

    char* ws = (char*)d_ws;
    size_t cur = 0;
    auto alloc = [&](size_t bytes) -> char* {
        char* p = ws + cur;
        cur = (cur + bytes + 255) & ~(size_t)255;
        return p;
    };

    float* x    = (float*)alloc((size_t)M_ROWS * D_MODEL * 4);
    bf16*  hb   = (bf16*) alloc((size_t)M_ROWS * D_MODEL * 2);
    bf16*  qbuf = (bf16*) alloc((size_t)N_IMG * NHEADS * S_LEN * HD * 2);
    bf16*  kbuf = (bf16*) alloc((size_t)N_IMG * NHEADS * S_LEN * HD * 2);
    bf16*  vbuf = (bf16*) alloc((size_t)N_IMG * NHEADS * S_LEN * HD * 2);
    bf16*  mlpb = (bf16*) alloc((size_t)M_ROWS * MLP_D * 2);
    bf16*  wqb  = (bf16*) alloc((size_t)NHEADS * HD * HD * 2);
    bf16*  wkb  = (bf16*) alloc((size_t)NHEADS * HD * HD * 2);
    bf16*  wvb  = (bf16*) alloc((size_t)NHEADS * HD * HD * 2);
    bf16*  w1b  = (bf16*) alloc((size_t)MLP_D * D_MODEL * 2);
    bf16*  w2b  = (bf16*) alloc((size_t)D_MODEL * MLP_D * 2);
    bf16*  wob  = (bf16*) alloc((size_t)OUT_PAD * D_MODEL * 2);
    bf16*  xcls = (bf16*) alloc((size_t)N_IMG * D_MODEL * 2);
    float* logi = (float*)alloc((size_t)N_IMG * OUT_PAD * 4);

    // 1) patch embed + cls + positions -> x
    {
        long tot = (long)M_ROWS * D_MODEL;
        patch_embed_kernel<<<(unsigned)((tot + 255) / 256), 256, 0, stream>>>(images, Wp, bp, cls, x);
    }

    const long wqkv_n = (long)NHEADS * HD * HD;          // 49152
    const long wmlp_n = (long)MLP_D * D_MODEL;           // 2359296

    for (int l = 0; l < L_LAYERS; ++l) {
        layernorm_kernel<<<M_ROWS, 256, 0, stream>>>(x, ln1_g + (long)l * D_MODEL, ln1_b + (long)l * D_MODEL, hb);
        cvt_bf16_kernel<<<(unsigned)((wqkv_n + 255) / 256), 256, 0, stream>>>(Wq + (long)l * wqkv_n, wqb, wqkv_n);
        cvt_bf16_kernel<<<(unsigned)((wqkv_n + 255) / 256), 256, 0, stream>>>(Wk + (long)l * wqkv_n, wkb, wqkv_n);
        cvt_bf16_kernel<<<(unsigned)((wqkv_n + 255) / 256), 256, 0, stream>>>(Wv + (long)l * wqkv_n, wvb, wqkv_n);
        qkv_wmma_kernel<<<dim3(M_ROWS / 16, NHEADS), 128, 0, stream>>>(
            hb, wqb, wkb, wvb,
            bq + (long)l * NHEADS * HD, bk + (long)l * NHEADS * HD, bv + (long)l * NHEADS * HD,
            qbuf, kbuf, vbuf);
        attn_wmma_kernel<<<N_IMG * NHEADS, 256, 0, stream>>>(qbuf, kbuf, vbuf, x);
        layernorm_kernel<<<M_ROWS, 256, 0, stream>>>(x, ln2_g + (long)l * D_MODEL, ln2_b + (long)l * D_MODEL, hb);
        cvt_bf16_kernel<<<(unsigned)((wmlp_n + 255) / 256), 256, 0, stream>>>(W1 + (long)l * wmlp_n, w1b, wmlp_n);
        cvt_bf16_kernel<<<(unsigned)((wmlp_n + 255) / 256), 256, 0, stream>>>(W2 + (long)l * wmlp_n, w2b, wmlp_n);
        // MLP1: gelu(hb @ W1^T + b1) -> mlpb (bf16)
        gemm_wmma_kernel<<<dim3(M_ROWS / 128, MLP_D / 64), 256, 0, stream>>>(
            hb, w1b, b1 + (long)l * MLP_D, nullptr, nullptr, mlpb,
            MLP_D, D_MODEL, MLP_D, 1);
        // MLP2: mlpb @ W2^T + b2, residual-add into x
        gemm_wmma_kernel<<<dim3(M_ROWS / 128, D_MODEL / 64), 256, 0, stream>>>(
            mlpb, w2b, b2 + (long)l * D_MODEL, x, nullptr, nullptr,
            D_MODEL, MLP_D, D_MODEL, 0);
    }

    // classifier head on class token
    extract_cls_kernel<<<(N_IMG * D_MODEL + 255) / 256, 256, 0, stream>>>(x, xcls);
    {
        long tot = (long)OUT_PAD * D_MODEL;
        cvt_pad_rows_kernel<<<(unsigned)((tot + 255) / 256), 256, 0, stream>>>(Wo, wob, OUT_N, OUT_PAD, D_MODEL);
    }
    gemm_wmma_kernel<<<dim3(N_IMG / 128, OUT_PAD / 64), 256, 0, stream>>>(
        xcls, wob, bo, nullptr, logi, nullptr,
        OUT_N, D_MODEL, OUT_PAD, 0);
    softmax_out_kernel<<<N_IMG, 256, 0, stream>>>(logi, (float*)d_out);
}
